// RelationalBertSelfAttention_83021717831829
// MI455X (gfx1250) — matrix operand order repaired and checked by reading, not compiled, not run on hardware
//
#include <hip/hip_runtime.h>

#define Bn   4
#define Sn   512
#define HIDn 768
#define NHn  12
#define DHn  64

typedef __attribute__((ext_vector_type(16))) __bf16        v16bf;
typedef __attribute__((ext_vector_type(8)))  float         v8f;
typedef __attribute__((ext_vector_type(4)))  unsigned int  u32x4;

// LDS layout for the fused kernel (dynamic shared memory)
#define RELLD    520                       // padded row (bf16 elems), 1040B rows (16B aligned)
#define SM_REL   (64 * RELLD * 2)          // relT[d][k] bf16 transposed rel tile
#define SM_SBUF  (12 * 512 * 4)            // scores rows (f32)
#define SM_PBUF  (16 * 512 * 2)            // probs rows (bf16, rows 12..15 zero pad)
#define SM_PART  (4 * 256 * 4)             // split-K partial tiles
#define SMEM_BYTES (SM_REL + SM_SBUF + SM_PBUF + SM_PART)   // 111616 B

union V16U {
    v16bf v;
    unsigned short u[16];
    u32x4 q[2];
};

__device__ __forceinline__ unsigned short f2bf(float f) {
    __bf16 h = (__bf16)f;                  // HW f32->bf16 convert (RNE)
    return __builtin_bit_cast(unsigned short, h);
}

__device__ __forceinline__ v8f wmma_bf16(v16bf a, v16bf b, v8f c) {
    return __builtin_amdgcn_wmma_f32_16x16x32_bf16(
        false, a, false, b, (short)0, c, false, false);
}

__device__ __forceinline__ v8f zero8() {
    v8f z = {0.f, 0.f, 0.f, 0.f, 0.f, 0.f, 0.f, 0.f};
    return z;
}

// ---- fragment loaders ------------------------------------------------------
// A (16x32 bf16): lane0-15 rows M=lane; element e -> K = e+16*(lane/16)
__device__ __forceinline__ v16bf load_a_f32(const float* rowp, int lane) {
    const float* p = rowp + ((lane >> 4) << 4);
    V16U o;
#pragma unroll
    for (int e = 0; e < 8; ++e) o.u[e] = f2bf(p[e]);
#pragma unroll
    for (int e = 0; e < 8; ++e) o.u[8 + e] = f2bf(p[8 + e]);
    return o.v;
}

__device__ __forceinline__ v16bf load_a_bf16(const unsigned short* rowp, int lane) {
    const unsigned short* p = rowp + ((lane >> 4) << 4);
    V16U o;
    o.q[0] = *(const u32x4*)(p);        // K = base + 0..7
    o.q[1] = *(const u32x4*)(p + 8);    // K = base + 8..15
    return o.v;
}

// B (32x16): col = lane%16, element e -> K = e + 16*(lane/16); K-contiguous source
__device__ __forceinline__ v16bf load_b_bf16_contig(const unsigned short* p) {
    V16U o;
    o.q[0] = *(const u32x4*)(p);
    o.q[1] = *(const u32x4*)(p + 8);
    return o.v;
}

// ---- kernel 0: Wt[mat][n][k] = bf16(W[k][n])  (32x32 LDS tile transpose) ---
__global__ void wt_kernel(const float* __restrict__ Wq, const float* __restrict__ Wk,
                          const float* __restrict__ Wv, unsigned short* __restrict__ Wt) {
    __shared__ float tile[32][33];
    int mat = blockIdx.z;
    const float* W = (mat == 0) ? Wq : (mat == 1) ? Wk : Wv;
    int n0 = blockIdx.x * 32, k0 = blockIdx.y * 32;
    int tx = threadIdx.x, ty = threadIdx.y;
    tile[ty][tx] = W[(size_t)(k0 + ty) * HIDn + n0 + tx];
    __syncthreads();
    Wt[((size_t)mat * HIDn + n0 + ty) * HIDn + k0 + tx] = f2bf(tile[tx][ty]);
}

// ---- kernel 1: QKV projection -> bf16 Q,K ([B,NH,S,DH]) and V^T ------------
__global__ void qkv_proj_kernel(const float* __restrict__ hid,
                                const unsigned short* __restrict__ Wt,
                                const float* __restrict__ bq,
                                const float* __restrict__ bk,
                                const float* __restrict__ bv,
                                unsigned short* __restrict__ Qbf,
                                unsigned short* __restrict__ Kbf,
                                unsigned short* __restrict__ Vt) {
    int wid  = blockIdx.x * (blockDim.x >> 5) + (threadIdx.x >> 5);
    int lane = threadIdx.x & 31;
    int mt  = wid & 127;
    int t2  = wid >> 7;
    int ng  = t2 % NHn;
    int mat = t2 / NHn;
    if (mat >= 3) return;
    const float* bias = (mat == 0) ? bq : (mat == 1) ? bk : bv;
    const unsigned short* Wm = Wt + (size_t)mat * HIDn * HIDn;

    int m0 = mt * 16, n0 = ng * 64;
    int arow = m0 + (lane & 15);
    v8f acc[4] = {zero8(), zero8(), zero8(), zero8()};

    for (int kk = 0; kk < HIDn; kk += 32) {
        v16bf a = load_a_f32(hid + (size_t)arow * HIDn + kk, lane);
#pragma unroll
        for (int t = 0; t < 4; ++t) {
            int col = n0 + t * 16 + (lane & 15);
            v16bf bf = load_b_bf16_contig(Wm + (size_t)col * HIDn + kk + ((lane >> 4) << 4));
            acc[t] = wmma_bf16(a, bf, acc[t]);
        }
    }
#pragma unroll
    for (int t = 0; t < 4; ++t) {
        int d = t * 16 + (lane & 15);
        float bias_v = bias[n0 + d];
#pragma unroll
        for (int r = 0; r < 8; ++r) {
            int m = m0 + r + ((lane >> 4) << 3);
            int b = m >> 9, s = m & 511;
            unsigned short h16 = f2bf(acc[t][r] + bias_v);
            if (mat == 2) {
                Vt[((size_t)(b * NHn + ng) * DHn + d) * Sn + s] = h16;
            } else {
                unsigned short* dst = (mat == 0) ? Qbf : Kbf;
                dst[((size_t)(b * NHn + ng) * Sn + s) * DHn + d] = h16;
            }
        }
    }
}

// ---- kernel 2: raw scores = q . k^T ----------------------------------------
__global__ void qk_scores_kernel(const unsigned short* __restrict__ Qbf,
                                 const unsigned short* __restrict__ Kbf,
                                 float* __restrict__ scores) {
    int wid  = blockIdx.x * (blockDim.x >> 5) + (threadIdx.x >> 5);
    int lane = threadIdx.x & 31;
    int kg = wid & 7;
    int qt = (wid >> 3) & 31;
    int bh = wid >> 8;
    if (bh >= Bn * NHn) return;
    const unsigned short* Qb = Qbf + (size_t)bh * Sn * DHn;
    const unsigned short* Kb = Kbf + (size_t)bh * Sn * DHn;
    int q0 = qt * 16, k0 = kg * 64;
    v8f acc[4] = {zero8(), zero8(), zero8(), zero8()};

#pragma unroll
    for (int dd = 0; dd < DHn; dd += 32) {
        v16bf a = load_a_bf16(Qb + (size_t)(q0 + (lane & 15)) * DHn + dd, lane);
#pragma unroll
        for (int t = 0; t < 4; ++t) {
            int k = k0 + t * 16 + (lane & 15);
            v16bf bf = load_b_bf16_contig(Kb + (size_t)k * DHn + dd + ((lane >> 4) << 4));
            acc[t] = wmma_bf16(a, bf, acc[t]);
        }
    }
    float* srow = scores + (size_t)bh * Sn * Sn;
#pragma unroll
    for (int t = 0; t < 4; ++t)
#pragma unroll
        for (int r = 0; r < 8; ++r) {
            int q = q0 + r + ((lane >> 4) << 3);
            int k = k0 + t * 16 + (lane & 15);
            srow[(size_t)q * Sn + k] = acc[t][r];
        }
}

// ---- kernel 3 (fused): per (b,q) workgroup of 8 waves ----------------------
//  phase 0: stage rel[b,q] (512x64 f32) -> LDS relT[d][k] bf16 (single HBM read)
//  phase A: scores_h = (qk + Q_heads . relT) * 1/8 + mask   -> LDS sbuf
//  phase B: softmax rows in LDS -> LDS pbuf + global probs
//  phase C: out = P_heads[16x512] . rel[512x64]  (split-K across waves)
__global__ void fused_rel_kernel(const float* __restrict__ rel,
                                 const unsigned short* __restrict__ Qbf,
                                 const float* __restrict__ qk,
                                 const float* __restrict__ mask,
                                 unsigned short* __restrict__ probs,
                                 float* __restrict__ out) {
    extern __shared__ __align__(16) char smem[];
    unsigned short* relT = (unsigned short*)smem;
    float*          sbuf = (float*)(smem + SM_REL);
    unsigned short* pbuf = (unsigned short*)(smem + SM_REL + SM_SBUF);
    float*          part = (float*)(smem + SM_REL + SM_SBUF + SM_PBUF);

    int tid  = threadIdx.x;
    int wave = tid >> 5, lane = tid & 31;
    int b = blockIdx.x >> 9, q = blockIdx.x & 511;
    const float* relbq = rel + (size_t)(b * Sn + q) * Sn * DHn;

    // phase 0: coalesced float4 reads, transposed bf16 scatter into LDS
    for (int i = tid; i < (Sn * DHn / 4); i += 256) {
        int k  = i >> 4;
        int d0 = (i & 15) * 4;
        float4 v = *(const float4*)(relbq + (size_t)k * DHn + d0);
        relT[(d0 + 0) * RELLD + k] = f2bf(v.x);
        relT[(d0 + 1) * RELLD + k] = f2bf(v.y);
        relT[(d0 + 2) * RELLD + k] = f2bf(v.z);
        relT[(d0 + 3) * RELLD + k] = f2bf(v.w);
    }
    for (int i = tid; i < 4 * Sn; i += 256) pbuf[12 * Sn + i] = 0;  // zero pad rows
    __syncthreads();

    // phase A: rel-score GEMM [12h x 64k] per wave (kg = wave), merge with qk
    {
        int k0 = wave * 64;
        int hr = lane & 15;
        int hc = hr < NHn ? hr : (NHn - 1);
        const unsigned short* arow = Qbf + ((size_t)(b * NHn + hc) * Sn + q) * DHn;
        v8f acc[4] = {zero8(), zero8(), zero8(), zero8()};
#pragma unroll
        for (int dd = 0; dd < DHn; dd += 32) {
            v16bf a = load_a_bf16(arow + dd, lane);
#pragma unroll
            for (int t = 0; t < 4; ++t) {
                int k = k0 + t * 16 + (lane & 15);
                // B[K=d][N=k] = relT[d][k]: stride RELLD between K elements (ds reads)
                const unsigned short* p = relT + (dd + ((lane >> 4) << 4)) * RELLD + k;
                V16U o;
#pragma unroll
                for (int e = 0; e < 16; ++e) o.u[e] = p[e * RELLD];
                acc[t] = wmma_bf16(a, o.v, acc[t]);
            }
        }
#pragma unroll
        for (int t = 0; t < 4; ++t)
#pragma unroll
            for (int r = 0; r < 8; ++r) {
                int h = r + ((lane >> 4) << 3);
                if (h < NHn) {
                    int k = k0 + t * 16 + (lane & 15);
                    float s = (qk[((size_t)(b * NHn + h) * Sn + q) * Sn + k] + acc[t][r])
                                  * 0.125f + mask[b * Sn + k];
                    sbuf[h * Sn + k] = s;
                }
            }
    }
    __syncthreads();

    // phase B: softmax of 12 rows in LDS
    for (int h = wave; h < NHn; h += 8) {
        float x[16];
        float mx = -3.0e38f;
#pragma unroll
        for (int e = 0; e < 16; ++e) {
            x[e] = sbuf[h * Sn + e * 32 + lane];
            mx = fmaxf(mx, x[e]);
        }
        for (int off = 16; off; off >>= 1) mx = fmaxf(mx, __shfl_xor(mx, off, 32));
        float sum = 0.f;
#pragma unroll
        for (int e = 0; e < 16; ++e) { x[e] = __expf(x[e] - mx); sum += x[e]; }
        for (int off = 16; off; off >>= 1) sum += __shfl_xor(sum, off, 32);
        float inv = 1.0f / sum;
        unsigned short* pg = probs + ((size_t)(b * NHn + h) * Sn + q) * Sn;
#pragma unroll
        for (int e = 0; e < 16; ++e) {
            unsigned short pb = f2bf(x[e] * inv);
            pbuf[h * Sn + e * 32 + lane] = pb;
            pg[e * 32 + lane] = pb;
        }
    }
    __syncthreads();

    // phase C: ctx_rel[16h x 64d] = pbuf[16x512] . relT^T, split-K over wave pairs
    {
        int t = wave & 3, half = wave >> 2;
        int d = t * 16 + (lane & 15);
        const unsigned short* arow = pbuf + (size_t)(lane & 15) * Sn;
        v8f acc = zero8();
        for (int kk = half * 256; kk < half * 256 + 256; kk += 32) {
            v16bf a = load_a_bf16(arow + kk, lane);                       // ds_load_b128
            v16bf bf = load_b_bf16_contig(relT + (size_t)d * RELLD + kk + ((lane >> 4) << 4));
            acc = wmma_bf16(a, bf, acc);
        }
        if (half) {
#pragma unroll
            for (int r = 0; r < 8; ++r) part[t * 256 + r * 32 + lane] = acc[r];
        }
        __syncthreads();
        if (!half) {
#pragma unroll
            for (int r = 0; r < 8; ++r) {
                int h = r + ((lane >> 4) << 3);
                if (h < NHn) {
                    float v = acc[r] + part[t * 256 + r * 32 + lane];
                    out[(size_t)(b * Sn + q) * HIDn + h * DHn + d] = v;  // '=' covers all of out
                }
            }
        }
    }
}

// ---- kernel 4: out += P[16x512] . V[512x64] --------------------------------
__global__ void pv_kernel(const unsigned short* __restrict__ probs,
                          const unsigned short* __restrict__ Vt,
                          float* __restrict__ out) {
    int wid  = blockIdx.x * (blockDim.x >> 5) + (threadIdx.x >> 5);
    int lane = threadIdx.x & 31;
    int qt = wid & 31;
    int bh = wid >> 5;
    if (bh >= Bn * NHn) return;
    int b = bh / NHn, h = bh % NHn;
    const unsigned short* prow = probs + ((size_t)bh * Sn + qt * 16 + (lane & 15)) * Sn;
    const unsigned short* Vb = Vt + (size_t)bh * DHn * Sn;
    v8f acc[4] = {zero8(), zero8(), zero8(), zero8()};

    for (int kk = 0; kk < Sn; kk += 32) {
        v16bf a = load_a_bf16(prow + kk, lane);
#pragma unroll
        for (int t = 0; t < 4; ++t) {
            int d = t * 16 + (lane & 15);
            v16bf bf = load_b_bf16_contig(Vb + (size_t)d * Sn + kk + ((lane >> 4) << 4));
            acc[t] = wmma_bf16(a, bf, acc[t]);
        }
    }
#pragma unroll
    for (int t = 0; t < 4; ++t)
#pragma unroll
        for (int r = 0; r < 8; ++r) {
            int q = qt * 16 + r + ((lane >> 4) << 3);
            int d = t * 16 + (lane & 15);
            out[(size_t)(b * Sn + q) * HIDn + h * DHn + d] += acc[t][r];
        }
}

extern "C" void kernel_launch(void* const* d_in, const int* in_sizes, int n_in,
                              void* d_out, int out_size, void* d_ws, size_t ws_size,
                              hipStream_t stream) {
    const float* hid  = (const float*)d_in[0];
    const float* mask = (const float*)d_in[1];
    const float* rel  = (const float*)d_in[2];
    const float* Wq   = (const float*)d_in[3];
    const float* bq   = (const float*)d_in[4];
    const float* Wk   = (const float*)d_in[5];
    const float* bk   = (const float*)d_in[6];
    const float* Wv   = (const float*)d_in[7];
    const float* bv   = (const float*)d_in[8];
    float* out = (float*)d_out;

    char* ws = (char*)d_ws;
    const size_t wtBytes     = (size_t)3 * HIDn * HIDn * sizeof(unsigned short);     // 3.5 MB
    const size_t qkvBytes    = (size_t)Bn * NHn * Sn * DHn * sizeof(unsigned short); // 3 MB
    const size_t scoresBytes = (size_t)Bn * NHn * Sn * Sn * sizeof(float);           // 48 MB
    unsigned short* Wt    = (unsigned short*)(ws);
    unsigned short* Qbf   = (unsigned short*)(ws + wtBytes);
    unsigned short* Kbf   = (unsigned short*)(ws + wtBytes + qkvBytes);
    unsigned short* Vt    = (unsigned short*)(ws + wtBytes + 2 * qkvBytes);
    float*          score = (float*)(ws + wtBytes + 3 * qkvBytes);
    unsigned short* probs = (unsigned short*)(ws + wtBytes + 3 * qkvBytes + scoresBytes);

    (void)hipFuncSetAttribute(reinterpret_cast<const void*>(fused_rel_kernel),
                              hipFuncAttributeMaxDynamicSharedMemorySize,
                              (int)SMEM_BYTES);

    const int TPB = 256, WPB = TPB / 32;
    wt_kernel<<<dim3(HIDn / 32, HIDn / 32, 3), dim3(32, 32), 0, stream>>>(Wq, Wk, Wv, Wt);
    qkv_proj_kernel<<<(3 * NHn * 128) / WPB, TPB, 0, stream>>>(
        hid, Wt, bq, bk, bv, Qbf, Kbf, Vt);
    qk_scores_kernel<<<(Bn * NHn * 32 * 8) / WPB, TPB, 0, stream>>>(Qbf, Kbf, score);
    fused_rel_kernel<<<Bn * Sn, TPB, SMEM_BYTES, stream>>>(
        rel, Qbf, score, mask, probs, out);
    pv_kernel<<<(Bn * NHn * 32) / WPB, TPB, 0, stream>>>(probs, Vt, out);
}
